// E_GCL_GKN_32452772888799
// MI455X (gfx1250) — matrix operand order repaired and checked by reading, not compiled, not run on hardware
//
#include <hip/hip_runtime.h>
#include <hip/hip_bf16.h>
#include <stdint.h>

typedef __attribute__((ext_vector_type(16))) _Float16 v16h;
typedef __attribute__((ext_vector_type(8)))  _Float16 v8h;
typedef __attribute__((ext_vector_type(8)))  float    v8f;
typedef __attribute__((ext_vector_type(4)))  int      v4i;

static constexpr int kN   = 10000;
static constexpr int kE   = 160000;
static constexpr int kHK  = 128;
static constexpr int kEA  = 6;

// ---------------- workspace layout (bytes) ----------------
// [0)       Wk2B : f16 planes, 2 * 65536 elems          = 262144 B
// [262144)  Bk2B : f16, 2ob * 32lane * 16               =   2048 B
// [264192)  Wc1B : f16 planes, 2 * 1024 elems           =   4096 B
// [268288)  sum_feat f32 kN*32 ; sum_trans f32 kN*3 ; deg f32 kN
static constexpr size_t kOffBk2 = 262144;
static constexpr size_t kOffWc1 = 264192;
static constexpr size_t kOffSF  = 268288;
static constexpr size_t kOffST  = kOffSF + (size_t)kN * 32 * 4;
static constexpr size_t kOffDeg = kOffST + (size_t)kN * 3 * 4;
static constexpr int kPackF16   = 134144;   // f16 elems in Wk2B|Bk2B|Wc1B
static constexpr int kPackBytes = 268288;

#if defined(__AMDGCN__) && __has_builtin(__builtin_amdgcn_global_load_async_to_lds_b128) && __has_builtin(__builtin_amdgcn_s_wait_asynccnt)
#define USE_ASYNC_COPY 1
#else
#define USE_ASYNC_COPY 0
#endif

__device__ __forceinline__ v16h joinh(v8h lo, v8h hi) {
  return __builtin_shufflevector(lo, hi, 0,1,2,3,4,5,6,7,8,9,10,11,12,13,14,15);
}

// ---------------------------------------------------------------------------
__global__ void zero_kernel(float* sum_feat, float* sum_trans, float* deg) {
  int idx = blockIdx.x * blockDim.x + threadIdx.x;
  if (idx < kN * 32)               sum_feat[idx] = 0.f;
  else if (idx < kN * 32 + kN * 3) sum_trans[idx - kN * 32] = 0.f;
  else if (idx < kN * 36)          deg[idx - kN * 35] = 0.f;
}

// ---------------------------------------------------------------------------
// Pack Wk2 (128x1024), bk2 (as 32x32) and Wc1 (32x64) into f16 WMMA B-operand
// tiles.  Wk2/Wc1 use a two-plane layout (8 f16 per lane per plane -> 16 B
// lane stride in LDS = conflict-minimal ds_load_b128).
__global__ void prepack_kernel(const float* __restrict__ Wk2,
                               const float* __restrict__ bk2,
                               const float* __restrict__ Wc1,
                               _Float16* __restrict__ Wk2B,
                               _Float16* __restrict__ Bk2B,
                               _Float16* __restrict__ Wc1B) {
  int idx = blockIdx.x * blockDim.x + threadIdx.x;
  if (idx < 131072) {                      // Wk2 planes
    int p = idx >> 16, rem = idx & 65535;
    int t8 = rem & 7, c = rem >> 3;
    int lane = c & 31, kt = (c >> 5) & 3, nb = c >> 7;
    int t = p * 8 + t8;
    int K = kt * 32 + p * 16 + (lane >> 4) * 8 + t8;
    int Nc = nb * 16 + (lane & 15);
    Wk2B[idx] = (_Float16)Wk2[(size_t)K * 1024 + Nc];
  } else if (idx < 132096) {               // bk2 tile (single-plane layout)
    int q = idx - 131072;
    int t = q & 15, lane = (q >> 4) & 31, ob = q >> 9;
    int K = (t >> 3) * 16 + (lane >> 4) * 8 + (t & 7);
    int Nc = ob * 16 + (lane & 15);
    Bk2B[q] = (_Float16)bk2[K * 32 + Nc];
  } else if (idx < kPackF16) {             // Wc1 planes
    int q = idx - 132096;
    int p = q >> 10, rem = q & 1023;
    int t8 = rem & 7, c = rem >> 3;
    int lane = c & 31, nb = c >> 5;
    int t = p * 8 + t8;
    int K = p * 16 + (lane >> 4) * 8 + t8;
    int Nc = nb * 16 + (lane & 15);
    Wc1B[q] = (_Float16)Wc1[K * 64 + Nc];
  }
}

// ---------------------------------------------------------------------------
__launch_bounds__(512, 1)
__global__ void edge_kernel(const float* __restrict__ h,
                            const int*   __restrict__ edge_index,
                            const float* __restrict__ coord,
                            const float* __restrict__ edge_attr,
                            const float* __restrict__ Wk1,
                            const float* __restrict__ bk1,
                            const float* __restrict__ bc1,
                            const float* __restrict__ Wc2,
                            const _Float16* __restrict__ pack_g,  // Wk2B|Bk2B|Wc1B
                            float* __restrict__ sum_feat,
                            float* __restrict__ sum_trans,
                            float* __restrict__ deg) {
  __shared__ __attribute__((aligned(16))) _Float16 sPack[kPackF16]; // 262 KB
  __shared__ float sWk1[kEA * kHK];
  __shared__ float sbk1[kHK];
  __shared__ float sbc1[64];
  __shared__ float sWc2[64];
  __shared__ float sScr[16][16 * 32];   // per-wave h-tile / feat-tile
  __shared__ int   sRow[16][16];
  __shared__ float sWcOut[16][16];

  _Float16* sB    = sPack;              // planes: [0,65536) lo, [65536,131072) hi
  _Float16* sBk2  = sPack + 131072;
  _Float16* sWc1B = sPack + 132096;     // planes: [0,1024) lo, [1024,2048) hi

  // ---- cooperative LDS fill (once per block) ----
#if USE_ASYNC_COPY
  {
    v4i* src = (v4i*)pack_g;
    v4i* dst = (v4i*)sPack;
    for (int i = threadIdx.x; i < kPackBytes / 16; i += blockDim.x)
      __builtin_amdgcn_global_load_async_to_lds_b128(src + i, dst + i, 0, 0);
    __builtin_amdgcn_s_wait_asynccnt(0);
  }
#else
  {
    const uint4* s = (const uint4*)pack_g;
    uint4* d = (uint4*)sPack;
    for (int i = threadIdx.x; i < kPackBytes / 16; i += blockDim.x) d[i] = s[i];
  }
#endif
  for (int i = threadIdx.x; i < kEA * kHK; i += blockDim.x) sWk1[i] = Wk1[i];
  for (int i = threadIdx.x; i < kHK; i += blockDim.x) sbk1[i] = bk1[i];
  for (int i = threadIdx.x; i < 64; i += blockDim.x) { sbc1[i] = bc1[i]; sWc2[i] = Wc2[i]; }
  __syncthreads();

  const int wave  = threadIdx.x >> 5;
  const int lane  = threadIdx.x & 31;
  const int m     = lane & 15;   // tile row / C-layout column
  const int khalf = lane >> 4;
  float* sH = sScr[wave];

  const int nTiles = kE / 16;
  for (int tile = blockIdx.x * 16 + wave; tile < nTiles; tile += gridDim.x * 16) {
    const int e = tile * 16 + m;
    const int r = edge_index[e];
    const int c = edge_index[kE + e];
    if (khalf == 0) sRow[wave][m] = r;

    // ---- A tiles: act = relu(edge_attr @ Wk1 + bk1), A-operand layout
    float eav[kEA];
    #pragma unroll
    for (int a = 0; a < kEA; ++a) eav[a] = edge_attr[(size_t)e * kEA + a];

    v16h A[4];
    #pragma unroll
    for (int kt = 0; kt < 4; ++kt) {
      #pragma unroll
      for (int t = 0; t < 8; ++t) {
        int K1 = kt * 32 + khalf * 8 + t;
        int K2 = K1 + 16;
        float v1 = sbk1[K1], v2 = sbk1[K2];
        #pragma unroll
        for (int a = 0; a < kEA; ++a) {
          v1 += eav[a] * sWk1[a * kHK + K1];
          v2 += eav[a] * sWk1[a * kHK + K2];
        }
        A[kt][t]     = (_Float16)fmaxf(v1, 0.f);
        A[kt][t + 8] = (_Float16)fmaxf(v2, 0.f);
      }
    }

    // ---- h[col] tile: A-layout f16 + f32 copy staged in LDS
    v16h Ah;
    #pragma unroll
    for (int t = 0; t < 8; ++t) {
      float v1 = h[(size_t)c * 32 + khalf * 8 + t];
      float v2 = h[(size_t)c * 32 + 16 + khalf * 8 + t];
      Ah[t] = (_Float16)v1;  Ah[t + 8] = (_Float16)v2;
      sH[m * 32 + khalf * 8 + t]      = v1;
      sH[m * 32 + 16 + khalf * 8 + t] = v2;
    }

    // ---- seed accumulators with the bk2 term (h_tile @ bk2-matrix)
    v8f zero = {};
    v16h b0 = *(const v16h*)(sBk2 + (0 * 32 + lane) * 16);
    v16h b1 = *(const v16h*)(sBk2 + (1 * 32 + lane) * 16);
    v8f acc0 = __builtin_amdgcn_wmma_f32_16x16x32_f16(false, Ah, false, b0,
                                                      (short)0, zero, false, false);
    v8f acc1 = __builtin_amdgcn_wmma_f32_16x16x32_f16(false, Ah, false, b1,
                                                      (short)0, zero, false, false);

    // ---- main GEMM: 64 blocks of 16 cols; nb -> (i = nb>>1, o-half = nb&1)
    for (int i = 0; i < 32; ++i) {
      v16h B0[4], B1[4];
      #pragma unroll
      for (int kt = 0; kt < 4; ++kt) {
        int c0 = ((2 * i) * 4 + kt) * 32 + lane;
        int c1 = ((2 * i + 1) * 4 + kt) * 32 + lane;
        B0[kt] = joinh(*(const v8h*)(sB + c0 * 8), *(const v8h*)(sB + 65536 + c0 * 8));
        B1[kt] = joinh(*(const v8h*)(sB + c1 * 8), *(const v8h*)(sB + 65536 + c1 * 8));
      }
      float hj[8];
      #pragma unroll
      for (int j = 0; j < 8; ++j) hj[j] = sH[(khalf * 8 + j) * 32 + i];

      v8f w0 = {}, w1 = {};
      #pragma unroll
      for (int kt = 0; kt < 4; ++kt) {
        w0 = __builtin_amdgcn_wmma_f32_16x16x32_f16(false, A[kt], false, B0[kt],
                                                    (short)0, w0, false, false);
        w1 = __builtin_amdgcn_wmma_f32_16x16x32_f16(false, A[kt], false, B1[kt],
                                                    (short)0, w1, false, false);
      }
      #pragma unroll
      for (int j = 0; j < 8; ++j) {
        acc0[j] += w0[j] * hj[j];
        acc1[j] += w1[j] * hj[j];
      }
    }

    // ---- edge_feat tile to LDS (row-major) + atomic scatter to nodes
    #pragma unroll
    for (int j = 0; j < 8; ++j) {
      int M = khalf * 8 + j;
      sH[M * 32 + m]      = acc0[j];
      sH[M * 32 + 16 + m] = acc1[j];
    }
    #pragma unroll
    for (int j = 0; j < 8; ++j) {
      int M = khalf * 8 + j;
      int rr = sRow[wave][M];
      atomicAdd(&sum_feat[(size_t)rr * 32 + m],      acc0[j]);
      atomicAdd(&sum_feat[(size_t)rr * 32 + 16 + m], acc1[j]);
    }

    // ---- coord MLP via WMMA: hidden = feat @ Wc1 + bc1 ; wc = relu(.)@Wc2
    v16h Af;
    #pragma unroll
    for (int t = 0; t < 8; ++t) {
      Af[t]     = (_Float16)sH[m * 32 + khalf * 8 + t];
      Af[t + 8] = (_Float16)sH[m * 32 + 16 + khalf * 8 + t];
    }
    float part[8];
    #pragma unroll
    for (int j = 0; j < 8; ++j) part[j] = 0.f;
    #pragma unroll
    for (int nb = 0; nb < 4; ++nb) {
      int u = nb * 16 + m;
      float bs = sbc1[u];
      v8f cs;
      #pragma unroll
      for (int j = 0; j < 8; ++j) cs[j] = bs;     // bias seeded through C operand
      int cc = nb * 32 + lane;
      v16h bw = joinh(*(const v8h*)(sWc1B + cc * 8),
                      *(const v8h*)(sWc1B + 1024 + cc * 8));
      v8f hid = __builtin_amdgcn_wmma_f32_16x16x32_f16(false, Af, false, bw,
                                                       (short)0, cs, false, false);
      float w2 = sWc2[u];
      #pragma unroll
      for (int j = 0; j < 8; ++j) part[j] += fmaxf(hid[j], 0.f) * w2;
    }
    #pragma unroll
    for (int j = 0; j < 8; ++j) {        // reduce over the 16 lanes of each half
      float p2 = part[j];
      p2 += __shfl_xor(p2, 1, 32);
      p2 += __shfl_xor(p2, 2, 32);
      p2 += __shfl_xor(p2, 4, 32);
      p2 += __shfl_xor(p2, 8, 32);
      part[j] = p2;
    }
    if (m == 0) {
      #pragma unroll
      for (int j = 0; j < 8; ++j) sWcOut[wave][khalf * 8 + j] = part[j];
    }
    float wc = sWcOut[wave][m];
    if (khalf == 0) {
      #pragma unroll
      for (int d = 0; d < 3; ++d) {
        float diff = coord[(size_t)r * 3 + d] - coord[(size_t)c * 3 + d];
        atomicAdd(&sum_trans[(size_t)r * 3 + d], diff * wc);
      }
      atomicAdd(&deg[r], 1.0f);
    }
  }
}

// ---------------------------------------------------------------------------
__global__ void node_kernel(const float* __restrict__ h,
                            const float* __restrict__ coord,
                            const float* __restrict__ root,
                            const float* __restrict__ bias,
                            const float* __restrict__ sum_feat,
                            const float* __restrict__ sum_trans,
                            const float* __restrict__ deg,
                            float* __restrict__ out) {
  int idx = blockIdx.x * blockDim.x + threadIdx.x;
  if (idx < kN * 32) {
    int n = idx >> 5, o = idx & 31;
    float d = fmaxf(deg[n], 1.0f);
    float agg = sum_feat[idx] / d + bias[o];
    const float* hr = h + (size_t)n * 32;
    float s = 0.f;
    #pragma unroll
    for (int i = 0; i < 32; ++i) s += hr[i] * root[i * 32 + o];
    agg += s;
    out[idx] = hr[o] + fmaxf(agg, 0.f) * 0.25f;
  } else if (idx < kN * 32 + kN * 3) {
    int k = idx - kN * 32;
    int n = k / 3;
    float d = fmaxf(deg[n], 1.0f);
    out[idx] = sum_trans[k] / d * 0.25f + coord[k];
  }
}

// ---------------------------------------------------------------------------
extern "C" void kernel_launch(void* const* d_in, const int* in_sizes, int n_in,
                              void* d_out, int out_size, void* d_ws, size_t ws_size,
                              hipStream_t stream) {
  (void)in_sizes; (void)n_in; (void)out_size; (void)ws_size;
  const float* h     = (const float*)d_in[0];
  const int*   eidx  = (const int*)d_in[1];
  const float* coord = (const float*)d_in[2];
  const float* eattr = (const float*)d_in[3];
  const float* Wk1   = (const float*)d_in[4];
  const float* bk1   = (const float*)d_in[5];
  const float* Wk2   = (const float*)d_in[6];
  const float* bk2   = (const float*)d_in[7];
  const float* root  = (const float*)d_in[8];
  const float* bias  = (const float*)d_in[9];
  const float* Wc1   = (const float*)d_in[10];
  const float* bc1   = (const float*)d_in[11];
  const float* Wc2   = (const float*)d_in[12];

  char* ws = (char*)d_ws;
  _Float16* Wk2B   = (_Float16*)ws;
  _Float16* Bk2B   = (_Float16*)(ws + kOffBk2);
  _Float16* Wc1B   = (_Float16*)(ws + kOffWc1);
  float* sum_feat  = (float*)(ws + kOffSF);
  float* sum_trans = (float*)(ws + kOffST);
  float* deg       = (float*)(ws + kOffDeg);
  float* out = (float*)d_out;

  {
    int total = kN * 36;
    zero_kernel<<<(total + 255) / 256, 256, 0, stream>>>(sum_feat, sum_trans, deg);
  }
  prepack_kernel<<<(kPackF16 + 255) / 256, 256, 0, stream>>>(Wk2, bk2, Wc1,
                                                             Wk2B, Bk2B, Wc1B);
  edge_kernel<<<256, 512, 0, stream>>>(h, eidx, coord, eattr, Wk1, bk1,
                                       bc1, Wc2, Wk2B,
                                       sum_feat, sum_trans, deg);
  {
    int total = kN * 32 + kN * 3;
    node_kernel<<<(total + 255) / 256, 256, 0, stream>>>(h, coord, root, bias,
                                                         sum_feat, sum_trans, deg, out);
  }
}
